// GQAttention_49091476193486
// MI455X (gfx1250) — compile-verified, matrix operand order checked by reading, not run on hardware
//
#include <hip/hip_runtime.h>

#define B_ 2
#define T_ 2048
#define D_ 1792
#define HQ_ 28
#define HKV_ 4
#define HD_ 64
#define GROUPS_ 7
#define NQKV 2304              // (HQ + 2*HKV) * HD
#define SCALE_ 0.125f          // 64^-0.5

typedef unsigned short u16;
typedef unsigned int   u32;
typedef __attribute__((ext_vector_type(16))) __bf16 v16bf;
typedef __attribute__((ext_vector_type(8)))  float  v8f;
typedef __attribute__((ext_vector_type(4)))  u32    v4u;
typedef __attribute__((ext_vector_type(8)))  int    v8i;
typedef __attribute__((ext_vector_type(4)))  int    v4i;

#ifndef __has_builtin
#define __has_builtin(x) 0
#endif
#if __has_builtin(__builtin_amdgcn_tensor_load_to_lds) && \
    __has_builtin(__builtin_amdgcn_s_wait_tensorcnt)
#define USE_TDM 1
#else
#define USE_TDM 0
#endif

union Frag { v16bf v; v4u q[2]; };

__device__ __forceinline__ u16 f2bf(float f) {
  u32 u = __builtin_bit_cast(u32, f);
  u32 r = u + 0x7FFFu + ((u >> 16) & 1u);   // round-to-nearest-even
  return (u16)(r >> 16);
}
__device__ __forceinline__ float bf2f(u16 h) {
  u32 u = ((u32)h) << 16;
  return __builtin_bit_cast(float, u);
}

#if USE_TDM
// Issue one TDM load of a [rows=128] x [32 bf16] tile (row stride = dim0
// elements) from global address ga into LDS at byte offset lds_addr.
// D# layout per CDNA5 ISA ch.8: group0 = {count, lds_addr, global_addr, type=2},
// group1 = {data_size=2B, tensor_dim0/1, tile_dim0=32, tile_dim1=128, dim0 stride}.
__device__ __forceinline__ void tdm_load_tile(u32 lds_addr, unsigned long long ga,
                                              int dim0, int rows) {
  u32 d0 = (u32)dim0, rr = (u32)rows;
  v4u g0 = (v4u){1u,                                   // count=1, user descriptor
                 lds_addr,                             // LDS byte address
                 (u32)ga,                              // global_addr[31:0]
                 ((u32)(ga >> 32) & 0x1FFFFFFu) | 0x80000000u};  // addr[56:32] | type=2
  v8i g1 = (v8i){(int)0x00010000u,                     // data_size=1 (2 bytes)
                 (int)((d0 & 0xFFFFu) << 16),          // tensor_dim0[15:0]
                 (int)((d0 >> 16) | ((rr & 0xFFFFu) << 16)),  // dim0 hi | dim1 lo
                 (int)((rr >> 16) | (32u << 16)),      // dim1 hi | tile_dim0=32
                 (int)128,                             // tile_dim1=128, tile_dim2=0
                 (int)d0,                              // tensor_dim0_stride[31:0]
                 0, 0};
  v4i z4 = (v4i){0, 0, 0, 0};
#if __clang_major__ >= 23
  v8i z8 = (v8i){0, 0, 0, 0, 0, 0, 0, 0};
  __builtin_amdgcn_tensor_load_to_lds(g0, g1, z4, z4, z8, 0);
#else
  __builtin_amdgcn_tensor_load_to_lds(g0, g1, z4, z4, 0);
#endif
}
#endif

// ---------------- conversion / packing kernels ----------------

__global__ void k_cast_x(const float* __restrict__ x, u16* __restrict__ xb, int n) {
  int i = blockIdx.x * blockDim.x + threadIdx.x;
  if (i < n) xb[i] = f2bf(x[i]);
}

// WT[n][k] = concat(Wq,Wk,Wv)[k][n], bf16, so GEMM B-fragments are K-contiguous
__global__ void k_pack_wqkv(const float* __restrict__ Wq, const float* __restrict__ Wk,
                            const float* __restrict__ Wv, u16* __restrict__ WT) {
  int i = blockIdx.x * blockDim.x + threadIdx.x;
  if (i >= NQKV * D_) return;
  int k = i % D_, n = i / D_;
  float val;
  if (n < HQ_ * HD_)                 val = Wq[(size_t)k * (HQ_ * HD_) + n];
  else if (n < (HQ_ + HKV_) * HD_)   val = Wk[(size_t)k * (HKV_ * HD_) + (n - HQ_ * HD_)];
  else                               val = Wv[(size_t)k * (HKV_ * HD_) + (n - (HQ_ + HKV_) * HD_)];
  WT[i] = f2bf(val);
}

__global__ void k_pack_bias(const float* __restrict__ bq, const float* __restrict__ bk,
                            const float* __restrict__ bv, float* __restrict__ bias) {
  int n = blockIdx.x * blockDim.x + threadIdx.x;
  if (n >= NQKV) return;
  float v;
  if (n < HQ_ * HD_)               v = bq[n];
  else if (n < (HQ_ + HKV_) * HD_) v = bk[n - HQ_ * HD_];
  else                             v = bv[n - (HQ_ + HKV_) * HD_];
  bias[n] = v;
}

__global__ void k_pack_wo(const float* __restrict__ Wo, u16* __restrict__ WoT) {
  int i = blockIdx.x * blockDim.x + threadIdx.x;
  if (i >= D_ * D_) return;
  int k = i % D_, n = i / D_;
  WoT[i] = f2bf(Wo[(size_t)k * D_ + n]);
}

// ---------------- bf16 WMMA GEMM: C[M][N] = A[M][K] * BT[N][K]^T + bias ----
// TDM double-buffered tile staging (wave 0 drives the Tensor Data Mover).

template <bool STORE_F32>
__global__ __launch_bounds__(256) void k_gemm(const u16* __restrict__ A,
                                              const u16* __restrict__ BT,
                                              const float* __restrict__ bias,
                                              void* __restrict__ Cout,
                                              int M, int N, int K) {
  __shared__ alignas(16) u16 lA[2][128 * 32];
  __shared__ alignas(16) u16 lB[2][128 * 32];
  const int tid = threadIdx.x;
  const int lane = tid & 31, wave = tid >> 5;
  const int lh = lane >> 4, l15 = lane & 15;      // half-wave, lane-in-half
  const int wm = wave >> 2, wn = wave & 3;        // 2 (M) x 4 (N) wave grid
  const int bm = blockIdx.x * 128, bn = blockIdx.y * 128;

  v8f acc[4][2];
#pragma unroll
  for (int i = 0; i < 4; ++i)
#pragma unroll
    for (int j = 0; j < 2; ++j) acc[i][j] = (v8f){0.f, 0.f, 0.f, 0.f, 0.f, 0.f, 0.f, 0.f};

  const int nst = K / 32;

#if USE_TDM
  const unsigned long long gAa = (unsigned long long)(uintptr_t)(A + (size_t)bm * K);
  const unsigned long long gBa = (unsigned long long)(uintptr_t)(BT + (size_t)bn * K);
  const u32 ldsA[2] = {(u32)(uintptr_t)&lA[0][0], (u32)(uintptr_t)&lA[1][0]};
  const u32 ldsB[2] = {(u32)(uintptr_t)&lB[0][0], (u32)(uintptr_t)&lB[1][0]};
  if (wave == 0) {
    tdm_load_tile(ldsA[0], gAa, K, M);
    tdm_load_tile(ldsB[0], gBa, K, N);
  }
#endif

  for (int i = 0; i < nst; ++i) {
    const int cur = i & 1;
#if USE_TDM
    if (wave == 0) {
      if (i + 1 < nst) {
        // prefetch next stage into the other buffer via the TDM
        tdm_load_tile(ldsA[cur ^ 1], gAa + (unsigned long long)(i + 1) * 64ull, K, M);
        tdm_load_tile(ldsB[cur ^ 1], gBa + (unsigned long long)(i + 1) * 64ull, K, N);
        __builtin_amdgcn_s_wait_tensorcnt((short)2);   // current pair landed
      } else {
        __builtin_amdgcn_s_wait_tensorcnt((short)0);
      }
    }
#else
    {
      const int k0 = i * 32;
#pragma unroll
      for (int t = 0; t < 2; ++t) {
        int qi = tid + t * 256;
        int row = qi >> 2, c4 = (qi & 3) * 8;
        *(v4u*)&lA[cur][row * 32 + c4] = *(const v4u*)&A[(size_t)(bm + row) * K + k0 + c4];
        *(v4u*)&lB[cur][row * 32 + c4] = *(const v4u*)&BT[(size_t)(bn + row) * K + k0 + c4];
        if (k0 + 32 < K) {
          __builtin_prefetch(&A[(size_t)(bm + row) * K + k0 + 32 + c4], 0, 3);
          __builtin_prefetch(&BT[(size_t)(bn + row) * K + k0 + 32 + c4], 0, 3);
        }
      }
    }
#endif
    __syncthreads();
    const u16* pA = &lA[cur][0];
    const u16* pB = &lB[cur][0];
    Frag fa[4], fb[2];
#pragma unroll
    for (int mt = 0; mt < 4; ++mt) {
      int m = wm * 64 + mt * 16 + l15;
      fa[mt].q[0] = *(const v4u*)&pA[m * 32 + lh * 8];
      fa[mt].q[1] = *(const v4u*)&pA[m * 32 + 16 + lh * 8];
    }
#pragma unroll
    for (int nt = 0; nt < 2; ++nt) {
      int n = wn * 32 + nt * 16 + l15;
      fb[nt].q[0] = *(const v4u*)&pB[n * 32 + lh * 8];
      fb[nt].q[1] = *(const v4u*)&pB[n * 32 + 16 + lh * 8];
    }
#pragma unroll
    for (int mt = 0; mt < 4; ++mt)
#pragma unroll
      for (int nt = 0; nt < 2; ++nt)
        acc[mt][nt] = __builtin_amdgcn_wmma_f32_16x16x32_bf16(
            false, fa[mt].v, false, fb[nt].v, (short)0, acc[mt][nt], false, false);
    __syncthreads();
  }

#pragma unroll
  for (int mt = 0; mt < 4; ++mt)
#pragma unroll
    for (int nt = 0; nt < 2; ++nt) {
      int gn = bn + wn * 32 + nt * 16 + l15;
      float bb = bias[gn];
#pragma unroll
      for (int r = 0; r < 8; ++r) {
        int gm = bm + wm * 64 + mt * 16 + lh * 8 + r;
        float v = acc[mt][nt][r] + bb;
        if (STORE_F32) ((float*)Cout)[(size_t)gm * N + gn] = v;
        else           ((u16*)Cout)[(size_t)gm * N + gn] = f2bf(v);
      }
    }
}

// ---------------- RoPE + per-head scatter ----------------

__global__ void k_rope_q(const u16* __restrict__ qkv, const float* __restrict__ cs,
                         const float* __restrict__ sn, u16* __restrict__ qh) {
  int i = blockIdx.x * blockDim.x + threadIdx.x;
  if (i >= B_ * T_ * HQ_ * 32) return;
  int d = i & 31;
  int h = (i >> 5) % HQ_;
  int t = ((i >> 5) / HQ_) % T_;
  int b = (i >> 5) / (HQ_ * T_);
  const u16* src = qkv + (size_t)(b * T_ + t) * NQKV + h * HD_;
  float x1 = bf2f(src[d]), x2 = bf2f(src[d + 32]);
  float c = cs[t * 32 + d], s = sn[t * 32 + d];
  u16* dst = qh + ((size_t)(b * HQ_ + h) * T_ + t) * HD_;
  dst[d]      = f2bf(x1 * c - x2 * s);
  dst[d + 32] = f2bf(x2 * c + x1 * s);
}

__global__ void k_rope_k(const u16* __restrict__ qkv, const float* __restrict__ cs,
                         const float* __restrict__ sn, u16* __restrict__ kh) {
  int i = blockIdx.x * blockDim.x + threadIdx.x;
  if (i >= B_ * T_ * HKV_ * 32) return;
  int d = i & 31;
  int h = (i >> 5) % HKV_;
  int t = ((i >> 5) / HKV_) % T_;
  int b = (i >> 5) / (HKV_ * T_);
  const u16* src = qkv + (size_t)(b * T_ + t) * NQKV + HQ_ * HD_ + h * HD_;
  float x1 = bf2f(src[d]), x2 = bf2f(src[d + 32]);
  float c = cs[t * 32 + d], s = sn[t * 32 + d];
  u16* dst = kh + ((size_t)(b * HKV_ + h) * T_ + t) * HD_;
  dst[d]      = f2bf(x1 * c - x2 * s);
  dst[d + 32] = f2bf(x2 * c + x1 * s);
}

// V scattered transposed: vT[b][h][d][t] so attention B-fragments are contiguous
__global__ void k_scat_v(const u16* __restrict__ qkv, u16* __restrict__ vT) {
  int i = blockIdx.x * blockDim.x + threadIdx.x;
  if (i >= B_ * T_ * HKV_ * HD_) return;
  int d = i & 63;
  int h = (i >> 6) % HKV_;
  int t = ((i >> 6) / HKV_) % T_;
  int b = (i >> 6) / (HKV_ * T_);
  u16 v = qkv[(size_t)(b * T_ + t) * NQKV + (HQ_ + HKV_) * HD_ + h * HD_ + d];
  vT[((size_t)(b * HKV_ + h) * HD_ + d) * T_ + t] = v;
}

// ---------------- causal GQA flash attention (WMMA bf16) ----------------

__global__ __launch_bounds__(128) void k_flash(const u16* __restrict__ qh,
                                               const u16* __restrict__ kh,
                                               const u16* __restrict__ vT,
                                               u16* __restrict__ ao) {
  __shared__ alignas(16) u16 lP[4][16 * 32];      // per-wave P transpose scratch
  const int lane = threadIdx.x & 31, wave = threadIdx.x >> 5;
  const int lh = lane >> 4, l15 = lane & 15;
  const int b = blockIdx.z, hq = blockIdx.y, hk = hq / GROUPS_;
  const int q0 = blockIdx.x * 64 + wave * 16;     // 16 q rows per wave

  const u16* qp = qh + ((size_t)(b * HQ_ + hq) * T_ + q0) * HD_;
  const u16* kp = kh + (size_t)(b * HKV_ + hk) * T_ * HD_;
  const u16* vp = vT + (size_t)(b * HKV_ + hk) * HD_ * T_;

  Frag aq[2];
#pragma unroll
  for (int kk = 0; kk < 2; ++kk) {
    aq[kk].q[0] = *(const v4u*)&qp[l15 * HD_ + kk * 32 + lh * 8];
    aq[kk].q[1] = *(const v4u*)&qp[l15 * HD_ + kk * 32 + 16 + lh * 8];
  }

  v8f o[4];
#pragma unroll
  for (int i = 0; i < 4; ++i) o[i] = (v8f){0.f, 0.f, 0.f, 0.f, 0.f, 0.f, 0.f, 0.f};
  float mi[8], li[8];
#pragma unroll
  for (int r = 0; r < 8; ++r) { mi[r] = -1e30f; li[r] = 0.f; }

  const int qmax = q0 + 15;
  for (int j0 = 0; j0 <= qmax; j0 += 32) {
    v8f s[2];
#pragma unroll
    for (int t2 = 0; t2 < 2; ++t2) {
      int n = j0 + t2 * 16 + l15;                 // kv row for this lane
      Frag fk[2];
#pragma unroll
      for (int kk = 0; kk < 2; ++kk) {
        fk[kk].q[0] = *(const v4u*)&kp[(size_t)n * HD_ + kk * 32 + lh * 8];
        fk[kk].q[1] = *(const v4u*)&kp[(size_t)n * HD_ + kk * 32 + 16 + lh * 8];
      }
      v8f z = (v8f){0.f, 0.f, 0.f, 0.f, 0.f, 0.f, 0.f, 0.f};
      z = __builtin_amdgcn_wmma_f32_16x16x32_bf16(false, aq[0].v, false, fk[0].v,
                                                  (short)0, z, false, false);
      z = __builtin_amdgcn_wmma_f32_16x16x32_bf16(false, aq[1].v, false, fk[1].v,
                                                  (short)0, z, false, false);
      s[t2] = z;
    }
#pragma unroll
    for (int r = 0; r < 8; ++r) {
      int qrow = q0 + lh * 8 + r;
      float v0 = (j0 + l15 <= qrow)      ? s[0][r] * SCALE_ : -1e30f;
      float v1 = (j0 + 16 + l15 <= qrow) ? s[1][r] * SCALE_ : -1e30f;
      float mx = fmaxf(v0, v1);
#pragma unroll
      for (int off = 1; off < 16; off <<= 1) mx = fmaxf(mx, __shfl_xor(mx, off, 32));
      float mnew = fmaxf(mi[r], mx);
      float alpha = __expf(mi[r] - mnew);
      mi[r] = mnew;
      float p0 = __expf(v0 - mnew);
      float p1 = __expf(v1 - mnew);
      lP[wave][(lh * 8 + r) * 32 + l15]      = f2bf(p0);
      lP[wave][(lh * 8 + r) * 32 + 16 + l15] = f2bf(p1);
      float rs = p0 + p1;
#pragma unroll
      for (int off = 1; off < 16; off <<= 1) rs += __shfl_xor(rs, off, 32);
      li[r] = li[r] * alpha + rs;
#pragma unroll
      for (int nt = 0; nt < 4; ++nt) o[nt][r] = o[nt][r] * alpha;
    }
    Frag ap;                                      // P as 16x32 A-fragment from LDS
    ap.q[0] = *(const v4u*)&lP[wave][l15 * 32 + lh * 8];
    ap.q[1] = *(const v4u*)&lP[wave][l15 * 32 + 16 + lh * 8];
#pragma unroll
    for (int nt = 0; nt < 4; ++nt) {
      int d = nt * 16 + l15;
      Frag fv;
      fv.q[0] = *(const v4u*)&vp[(size_t)d * T_ + j0 + lh * 8];
      fv.q[1] = *(const v4u*)&vp[(size_t)d * T_ + j0 + 16 + lh * 8];
      o[nt] = __builtin_amdgcn_wmma_f32_16x16x32_bf16(false, ap.v, false, fv.v,
                                                      (short)0, o[nt], false, false);
    }
  }

#pragma unroll
  for (int nt = 0; nt < 4; ++nt) {
    int d = nt * 16 + l15;
#pragma unroll
    for (int r = 0; r < 8; ++r) {
      int t = q0 + lh * 8 + r;
      float val = o[nt][r] / li[r];
      ao[(size_t)(b * T_ + t) * (HQ_ * HD_) + hq * HD_ + d] = f2bf(val);
    }
  }
}

// ---------------- launch ----------------

extern "C" void kernel_launch(void* const* d_in, const int* in_sizes, int n_in,
                              void* d_out, int out_size, void* d_ws, size_t ws_size,
                              hipStream_t stream) {
  const float* x  = (const float*)d_in[0];
  const float* cs = (const float*)d_in[1];
  const float* sn = (const float*)d_in[2];
  const float* Wq = (const float*)d_in[3];
  const float* bq = (const float*)d_in[4];
  const float* Wk = (const float*)d_in[5];
  const float* bk = (const float*)d_in[6];
  const float* Wv = (const float*)d_in[7];
  const float* bv = (const float*)d_in[8];
  const float* Wo = (const float*)d_in[9];
  const float* bo = (const float*)d_in[10];
  float* out = (float*)d_out;

  char* ws = (char*)d_ws;
  size_t off = 0;
  auto alloc = [&](size_t bytes) -> char* {
    char* p = ws + off;
    off = (off + bytes + 255) & ~(size_t)255;
    return p;
  };
  const size_t M = (size_t)B_ * T_;                       // 4096
  u16*   xb    = (u16*)alloc(M * D_ * 2);                 // x in bf16
  u16*   WT    = (u16*)alloc((size_t)NQKV * D_ * 2);      // qkv weights, [N][K] bf16
  float* biasc = (float*)alloc((size_t)NQKV * 4);         // concat bias
  u16*   WoT   = (u16*)alloc((size_t)D_ * D_ * 2);        // Wo^T bf16
  u16*   qkv   = (u16*)alloc(M * NQKV * 2);               // raw projections bf16
  u16*   qh    = (u16*)alloc((size_t)B_ * HQ_ * T_ * HD_ * 2);
  u16*   khp   = (u16*)alloc((size_t)B_ * HKV_ * T_ * HD_ * 2);
  u16*   vTp   = (u16*)alloc((size_t)B_ * HKV_ * HD_ * T_ * 2);
  u16*   aob   = (u16*)alloc(M * D_ * 2);                 // attention out bf16
  (void)ws_size; (void)in_sizes; (void)n_in; (void)out_size;

  int n;
  n = B_ * T_ * D_;
  k_cast_x<<<(n + 255) / 256, 256, 0, stream>>>(x, xb, n);
  n = NQKV * D_;
  k_pack_wqkv<<<(n + 255) / 256, 256, 0, stream>>>(Wq, Wk, Wv, WT);
  k_pack_bias<<<(NQKV + 255) / 256, 256, 0, stream>>>(bq, bk, bv, biasc);
  n = D_ * D_;
  k_pack_wo<<<(n + 255) / 256, 256, 0, stream>>>(Wo, WoT);

  dim3 g1((int)(M / 128), NQKV / 128);                    // 32 x 18
  k_gemm<false><<<g1, 256, 0, stream>>>(xb, WT, biasc, qkv, (int)M, NQKV, D_);

  n = B_ * T_ * HQ_ * 32;
  k_rope_q<<<(n + 255) / 256, 256, 0, stream>>>(qkv, cs, sn, qh);
  n = B_ * T_ * HKV_ * 32;
  k_rope_k<<<(n + 255) / 256, 256, 0, stream>>>(qkv, cs, sn, khp);
  n = B_ * T_ * HKV_ * HD_;
  k_scat_v<<<(n + 255) / 256, 256, 0, stream>>>(qkv, vTp);

  dim3 g2(T_ / 64, HQ_, B_);                              // 32 x 28 x 2
  k_flash<<<g2, 128, 0, stream>>>(qh, khp, vTp, aob);

  dim3 g3((int)(M / 128), D_ / 128);                      // 32 x 14
  k_gemm<true><<<g3, 256, 0, stream>>>(aob, WoT, bo, out, (int)M, D_, D_);
}
